// ChannelSelfAttention_89395449299763
// MI455X (gfx1250) — matrix operand order repaired and checked by reading, not compiled, not run on hardware
//
#include <hip/hip_runtime.h>
#include <hip/hip_bf16.h>

typedef __bf16 bf16;
typedef __attribute__((ext_vector_type(16))) __bf16 v16bf;
typedef __attribute__((ext_vector_type(8)))  __bf16 v8bf;
typedef __attribute__((ext_vector_type(4)))  __bf16 v4bf;
typedef __attribute__((ext_vector_type(8)))  float  v8f;
typedef __attribute__((ext_vector_type(4)))  int    v4i;

constexpr int BATCH = 8;
constexpr int NTOK  = 4096;
constexpr int C     = 512;
constexpr int H     = 8;
constexpr int D     = 64;
constexpr int MTOK  = BATCH * NTOK;   // 32768
constexpr int C3    = 3 * C;          // 1536
constexpr int KCH   = 8;              // K-chunks for the scores reduction
constexpr int KCLEN = NTOK / KCH;     // 512
constexpr int BROW  = 40;             // padded LDS row stride (elements, 80B)

#if defined(__has_builtin)
#if __has_builtin(__builtin_amdgcn_global_load_async_to_lds_b128) && \
    __has_builtin(__builtin_amdgcn_s_wait_asynccnt)
#define HAVE_ASYNC_LDS 1
#endif
#endif
#ifndef HAVE_ASYNC_LDS
#define HAVE_ASYNC_LDS 0
#endif

// ---------------------------------------------------------------------------
// WMMA fragment helpers (16x16x32 bf16, wave32; contiguous k order used by this code for A and B)
// A (16x32): lane<16 -> row M=lane, K=0..15; lane>=16 -> row M=lane-16,
//            K=16..31. One contiguous 32B chunk per lane.
// B (32x16): lane -> column N=lane&15; lane<16 K=0..15, lane>=16 K=16..31.
//            One contiguous 32B chunk of the column (feed B transposed).
// C/D (16x16 f32): lane<16 N=lane, VGPR r -> M=r; lane>=16 N=lane-16, M=8+r.
// ---------------------------------------------------------------------------
__device__ __forceinline__ v16bf load_frag_a(const bf16* __restrict__ A, int lda,
                                             int row0, int k0, int lane) {
  int r  = lane & 15;
  int kb = (lane < 16) ? 0 : 16;
  const bf16* p = A + (size_t)(row0 + r) * lda + (k0 + kb);
  v8bf lo = *(const v8bf*)(p);
  v8bf hi = *(const v8bf*)(p + 8);
  return __builtin_shufflevector(lo, hi, 0,1,2,3,4,5,6,7,8,9,10,11,12,13,14,15);
}

__device__ __forceinline__ v16bf load_frag_bt(const bf16* __restrict__ Bt, int ldb,
                                              int col0, int k0, int lane) {
  int n  = col0 + (lane & 15);
  int ks = (lane < 16) ? 0 : 16;
  const bf16* p = Bt + (size_t)n * ldb + (k0 + ks);
  v8bf lo = *(const v8bf*)(p);
  v8bf hi = *(const v8bf*)(p + 8);
  return __builtin_shufflevector(lo, hi, 0,1,2,3,4,5,6,7,8,9,10,11,12,13,14,15);
}

// B fragment from a staged LDS tile (64 rows x 32 k, padded row stride BROW).
__device__ __forceinline__ v16bf load_frag_b_lds(const bf16* __restrict__ lbuf,
                                                 int j, int lane) {
  const bf16* p = lbuf + (16 * j + (lane & 15)) * BROW + ((lane < 16) ? 0 : 16);
  v8bf lo = *(const v8bf*)(p);
  v8bf hi = *(const v8bf*)(p + 8);
  return __builtin_shufflevector(lo, hi, 0,1,2,3,4,5,6,7,8,9,10,11,12,13,14,15);
}

__device__ __forceinline__ v8f wmma_bf16(v16bf a, v16bf b, v8f c) {
  return __builtin_amdgcn_wmma_f32_16x16x32_bf16(false, a, false, b,
                                                 (short)0, c, false, false);
}

// ---------------------------------------------------------------------------
// Stage a 64(n) x 32(k) bf16 weight tile into LDS: 256 threads x 16B each.
// Uses gfx1250 async DMA-to-LDS when the builtin exists.
// ---------------------------------------------------------------------------
__device__ __forceinline__ void stage_b_tile(const bf16* __restrict__ Bt, int ldb,
                                             int n0, int k0, bf16* __restrict__ lbuf,
                                             int tid) {
  const int n  = tid >> 2;        // 0..63
  const int ch = tid & 3;         // 16B chunk within the 64B row
  const bf16* g = Bt + (size_t)(n0 + n) * ldb + k0 + ch * 8;
  bf16* l = lbuf + n * BROW + ch * 8;
#if HAVE_ASYNC_LDS
  __builtin_amdgcn_global_load_async_to_lds_b128(
      (__attribute__((address_space(1))) v4i*)(v4i*)g,
      (__attribute__((address_space(3))) v4i*)(v4i*)l, 0, 0);
#else
  *(v8bf*)l = *(const v8bf*)g;
#endif
}

// ---------------------------------------------------------------------------
// Block-cooperative GEMM: 8 waves, block tile 256(M) x 64(N); each wave owns
// 32 rows. B tiles staged in double-buffered LDS (async), A double-buffered
// in registers.
// ---------------------------------------------------------------------------
template <int KITER>
__device__ __forceinline__ void gemm_block_stagedB(const bf16* __restrict__ A, int lda,
                                                   int m0,
                                                   const bf16* __restrict__ Bt, int ldb,
                                                   int n0,
                                                   bf16* __restrict__ sm0,
                                                   bf16* __restrict__ sm1,
                                                   int tid, int lane, v8f (&acc)[2][4]) {
  stage_b_tile(Bt, ldb, n0, 0, sm0, tid);
  v16bf a0 = load_frag_a(A, lda, m0,      0, lane);
  v16bf a1 = load_frag_a(A, lda, m0 + 16, 0, lane);

#pragma unroll
  for (int ks = 0; ks < KITER; ++ks) {
    bf16* cur = (ks & 1) ? sm1 : sm0;
    bf16* nxt = (ks & 1) ? sm0 : sm1;
    __syncthreads();                            // stage ks-1 readers done with nxt
    if (ks + 1 < KITER) stage_b_tile(Bt, ldb, n0, (ks + 1) * 32, nxt, tid);
#if HAVE_ASYNC_LDS
    if (ks + 1 < KITER) __builtin_amdgcn_s_wait_asynccnt(1);   // cur tile landed
    else                __builtin_amdgcn_s_wait_asynccnt(0);
#endif
    __syncthreads();                            // all waves see cur tile
    v16bf na0 = a0, na1 = a1;
    if (ks + 1 < KITER) {                       // prefetch next A frags
      na0 = load_frag_a(A, lda, m0,      (ks + 1) * 32, lane);
      na1 = load_frag_a(A, lda, m0 + 16, (ks + 1) * 32, lane);
    }
    // Issue all B-fragment LDS loads up front so DScnt waits can be relaxed.
    v16bf bf0 = load_frag_b_lds(cur, 0, lane);
    v16bf bf1 = load_frag_b_lds(cur, 1, lane);
    v16bf bf2 = load_frag_b_lds(cur, 2, lane);
    v16bf bf3 = load_frag_b_lds(cur, 3, lane);
    acc[0][0] = wmma_bf16(a0, bf0, acc[0][0]);
    acc[1][0] = wmma_bf16(a1, bf0, acc[1][0]);
    acc[0][1] = wmma_bf16(a0, bf1, acc[0][1]);
    acc[1][1] = wmma_bf16(a1, bf1, acc[1][1]);
    acc[0][2] = wmma_bf16(a0, bf2, acc[0][2]);
    acc[1][2] = wmma_bf16(a1, bf2, acc[1][2]);
    acc[0][3] = wmma_bf16(a0, bf3, acc[0][3]);
    acc[1][3] = wmma_bf16(a1, bf3, acc[1][3]);
    a0 = na0; a1 = na1;
  }
}

// ---------------------------------------------------------------------------
// Register-only 32x64 wave tile (for the attention-side GEMMs).
// ---------------------------------------------------------------------------
template <int KITER>
__device__ __forceinline__ void gemm_tile_32x64(const bf16* __restrict__ A, int lda,
                                                int m0, int k0a,
                                                const bf16* __restrict__ Bt, int ldb,
                                                int n0, int k0b,
                                                int lane, v8f (&acc)[2][4]) {
  v16bf a0 = load_frag_a(A, lda, m0,      k0a, lane);
  v16bf a1 = load_frag_a(A, lda, m0 + 16, k0a, lane);
  v16bf b0 = load_frag_bt(Bt, ldb, n0,      k0b, lane);
  v16bf b1 = load_frag_bt(Bt, ldb, n0 + 16, k0b, lane);
  v16bf b2 = load_frag_bt(Bt, ldb, n0 + 32, k0b, lane);
  v16bf b3 = load_frag_bt(Bt, ldb, n0 + 48, k0b, lane);
#pragma unroll
  for (int it = 0; it < KITER; ++it) {
    v16bf na0 = a0, na1 = a1, nb0 = b0, nb1 = b1, nb2 = b2, nb3 = b3;
    if (it + 1 < KITER) {
      const int ka = k0a + (it + 1) * 32;
      const int kb = k0b + (it + 1) * 32;
      na0 = load_frag_a(A, lda, m0,      ka, lane);
      na1 = load_frag_a(A, lda, m0 + 16, ka, lane);
      nb0 = load_frag_bt(Bt, ldb, n0,      kb, lane);
      nb1 = load_frag_bt(Bt, ldb, n0 + 16, kb, lane);
      nb2 = load_frag_bt(Bt, ldb, n0 + 32, kb, lane);
      nb3 = load_frag_bt(Bt, ldb, n0 + 48, kb, lane);
    }
    acc[0][0] = wmma_bf16(a0, b0, acc[0][0]);
    acc[1][0] = wmma_bf16(a1, b0, acc[1][0]);
    acc[0][1] = wmma_bf16(a0, b1, acc[0][1]);
    acc[1][1] = wmma_bf16(a1, b1, acc[1][1]);
    acc[0][2] = wmma_bf16(a0, b2, acc[0][2]);
    acc[1][2] = wmma_bf16(a1, b2, acc[1][2]);
    acc[0][3] = wmma_bf16(a0, b3, acc[0][3]);
    acc[1][3] = wmma_bf16(a1, b3, acc[1][3]);
    a0 = na0; a1 = na1; b0 = nb0; b1 = nb1; b2 = nb2; b3 = nb3;
  }
}

// ---------------------------------------------------------------------------
// Prep kernels
// ---------------------------------------------------------------------------
__global__ void cvt_f32_bf16_x4(const float* __restrict__ src, bf16* __restrict__ dst, int n) {
  int i = (blockIdx.x * 256 + threadIdx.x) * 4;
  if (i < n) {
    float4 f = *(const float4*)(src + i);
    v4bf o;
    o[0] = (bf16)f.x; o[1] = (bf16)f.y; o[2] = (bf16)f.z; o[3] = (bf16)f.w;
    *(v4bf*)(dst + i) = o;
  }
}

// wt[n][k] = w[k][n];  w is K x Ncols row-major, wt is Ncols x K row-major.
__global__ void transpose_w_bf16(const float* __restrict__ w, bf16* __restrict__ wt,
                                 int K, int Ncols) {
  int i = blockIdx.x * 256 + threadIdx.x;
  if (i < K * Ncols) {
    int n = i / K, k = i % K;
    wt[i] = (bf16)w[(size_t)k * Ncols + n];
  }
}

// ---------------------------------------------------------------------------
// GEMM1: qkv = x @ w_qkv + b_qkv.  Block tile 256x64 (LDS-staged weights).
// q,k -> channel-major qT/kT[(b*H+h)*D+d][n], v -> token-major v[t][c].
// Grid: 128 m-blocks x 24 n-tiles = 3072 blocks.
// ---------------------------------------------------------------------------
__global__ __launch_bounds__(256)
void gemm_qkv(const bf16* __restrict__ xb, const bf16* __restrict__ wqkvt,
              const float* __restrict__ bias,
              bf16* __restrict__ qT, bf16* __restrict__ kT, bf16* __restrict__ vmat) {
  __shared__ bf16 smem[2][64 * BROW];
  const int tid  = threadIdx.x;
  const int lane = tid & 31;
  const int nTiles = C3 / 64;                   // 24
  const int n0 = (blockIdx.x % nTiles) * 64;
  const int m0 = (blockIdx.x / nTiles) * 256 + (tid >> 5) * 32;

  v8f acc[2][4] = {};
  gemm_block_stagedB<C / 32>(xb, C, m0, wqkvt, C, n0, smem[0], smem[1], tid, lane, acc);

  const int rbase = (lane < 16) ? 0 : 8;
#pragma unroll
  for (int mi = 0; mi < 2; ++mi) {
    const int m0s = m0 + mi * 16;
    const int b_  = m0s >> 12;
    const int nn0 = (m0s & (NTOK - 1)) + rbase;
#pragma unroll
    for (int j = 0; j < 4; ++j) {
      const int col  = n0 + 16 * j + (lane & 15);
      const float bc = bias[col];
      if (col < 2 * C) {                        // q or k -> channel-major
        const int cc = (col < C) ? col : (col - C);
        bf16* dst = (col < C) ? qT : kT;
        const int h = cc >> 6, d = cc & 63;
        v8bf pack;
#pragma unroll
        for (int r = 0; r < 8; ++r) pack[r] = (bf16)(acc[mi][j][r] + bc);
        *(v8bf*)(dst + ((size_t)((b_ * H + h) * D + d)) * NTOK + nn0) = pack;
      } else {                                  // v -> token-major
        const int cc = col - 2 * C;
#pragma unroll
        for (int r = 0; r < 8; ++r)
          vmat[(size_t)(m0s + rbase + r) * C + cc] = (bf16)(acc[mi][j][r] + bc);
      }
    }
  }
}

// ---------------------------------------------------------------------------
// Partial scores: part[kc][bh][d][e] = sum_{n in chunk} qT[d,n]*kT[e,n]  (f32)
// 64 bh x 8 kchunks x 2 m-tiles = 1024 waves.
// ---------------------------------------------------------------------------
__global__ __launch_bounds__(256)
void scores_partial(const bf16* __restrict__ qT, const bf16* __restrict__ kT,
                    float* __restrict__ part) {
  const int lane = threadIdx.x & 31;
  const int gw = blockIdx.x * 8 + (threadIdx.x >> 5);
  const int mt = gw & 1;
  const int kc = (gw >> 1) & (KCH - 1);
  const int bh = gw >> 4;
  const int m0 = mt * 32;
  const int k0 = kc * KCLEN;
  const bf16* qb = qT + (size_t)bh * D * NTOK;
  const bf16* kb = kT + (size_t)bh * D * NTOK;

  v8f acc[2][4] = {};
  gemm_tile_32x64<KCLEN / 32>(qb, NTOK, m0, k0, kb, NTOK, 0, k0, lane, acc);

  const int rbase = (lane < 16) ? 0 : 8;
  float* out = part + (((size_t)kc * (BATCH * H) + bh) << 12);   // *4096
#pragma unroll
  for (int mi = 0; mi < 2; ++mi)
#pragma unroll
    for (int j = 0; j < 4; ++j) {
      const int e = 16 * j + (lane & 15);
#pragma unroll
      for (int r = 0; r < 8; ++r)
        out[(m0 + mi * 16 + rbase + r) * D + e] = acc[mi][j][r];
    }
}

// ---------------------------------------------------------------------------
// Reduce partials, apply temperature, softmax over e, emit bf16 attn[bh][d][e].
// ---------------------------------------------------------------------------
__global__ __launch_bounds__(64)
void softmax_attn(const float* __restrict__ part, const float* __restrict__ temp,
                  bf16* __restrict__ attn) {
  const int bh = blockIdx.x;
  const int d  = threadIdx.x;
  const float t = temp[bh & (H - 1)];
  const float* base = part + ((size_t)bh << 12) + d * D;

  float row[D];
#pragma unroll
  for (int e = 0; e < D; ++e) {
    float s = 0.f;
#pragma unroll
    for (int kc = 0; kc < KCH; ++kc)
      s += base[((size_t)kc * (BATCH * H) << 12) + e];
    row[e] = s * t;
  }
  float m = row[0];
#pragma unroll
  for (int e = 1; e < D; ++e) m = fmaxf(m, row[e]);
  float sum = 0.f;
#pragma unroll
  for (int e = 0; e < D; ++e) { row[e] = __expf(row[e] - m); sum += row[e]; }
  const float inv = 1.0f / sum;
  bf16* out = attn + (size_t)bh * D * D + d * D;
#pragma unroll
  for (int e = 0; e < D; ++e) out[e] = (bf16)(row[e] * inv);
}

// ---------------------------------------------------------------------------
// y[t, h*D+d] = sum_e v[t, h*D+e] * attn[bh][d][e]   (V @ attn^T)
// 64 bh x 128 token-tiles = 8192 waves.
// ---------------------------------------------------------------------------
__global__ __launch_bounds__(256)
void attn_apply(const bf16* __restrict__ vmat, const bf16* __restrict__ attn,
                bf16* __restrict__ y) {
  const int lane = threadIdx.x & 31;
  const int gw = blockIdx.x * 8 + (threadIdx.x >> 5);
  const int bh = gw >> 7;
  const int mt = gw & 127;
  const int b_ = bh >> 3, h = bh & (H - 1);
  const int tok0 = b_ * NTOK + mt * 32;
  const bf16* Bt = attn + (size_t)bh * D * D;

  v8f acc[2][4] = {};
  gemm_tile_32x64<D / 32>(vmat, C, tok0, h * D, Bt, D, 0, 0, lane, acc);

  const int rbase = (lane < 16) ? 0 : 8;
#pragma unroll
  for (int mi = 0; mi < 2; ++mi)
#pragma unroll
    for (int j = 0; j < 4; ++j) {
      const int col = h * D + 16 * j + (lane & 15);
#pragma unroll
      for (int r = 0; r < 8; ++r)
        y[(size_t)(tok0 + mi * 16 + rbase + r) * C + col] = (bf16)acc[mi][j][r];
    }
}

// ---------------------------------------------------------------------------
// out = y @ w_p + b_p (f32).  Block tile 256x64 (LDS-staged weights).
// Grid: 128 m-blocks x 8 n-tiles = 1024 blocks.
// ---------------------------------------------------------------------------
__global__ __launch_bounds__(256)
void gemm_out(const bf16* __restrict__ y, const bf16* __restrict__ wpt,
              const float* __restrict__ bias, float* __restrict__ out) {
  __shared__ bf16 smem[2][64 * BROW];
  const int tid  = threadIdx.x;
  const int lane = tid & 31;
  const int n0 = (blockIdx.x & 7) * 64;
  const int m0 = (blockIdx.x >> 3) * 256 + (tid >> 5) * 32;

  v8f acc[2][4] = {};
  gemm_block_stagedB<C / 32>(y, C, m0, wpt, C, n0, smem[0], smem[1], tid, lane, acc);

  const int rbase = (lane < 16) ? 0 : 8;
#pragma unroll
  for (int mi = 0; mi < 2; ++mi)
#pragma unroll
    for (int j = 0; j < 4; ++j) {
      const int col  = n0 + 16 * j + (lane & 15);
      const float bc = bias[col];
#pragma unroll
      for (int r = 0; r < 8; ++r)
        out[(size_t)(m0 + mi * 16 + rbase + r) * C + col] = acc[mi][j][r] + bc;
    }
}

// ---------------------------------------------------------------------------
extern "C" void kernel_launch(void* const* d_in, const int* in_sizes, int n_in,
                              void* d_out, int out_size, void* d_ws, size_t ws_size,
                              hipStream_t stream) {
  const float* x     = (const float*)d_in[0];
  const float* w_qkv = (const float*)d_in[1];
  const float* b_qkv = (const float*)d_in[2];
  const float* w_p   = (const float*)d_in[3];
  const float* b_p   = (const float*)d_in[4];
  const float* temp  = (const float*)d_in[5];

  char* ws = (char*)d_ws;
  size_t off = 0;
  auto carve = [&](size_t bytes) {
    char* p = ws + off;
    off += (bytes + 255) & ~(size_t)255;
    return p;
  };
  bf16*  xb    = (bf16*)carve((size_t)MTOK * C * 2);              // 32 MB
  bf16*  wqkvt = (bf16*)carve((size_t)C3 * C * 2);                // 1.5 MB
  bf16*  wpt   = (bf16*)carve((size_t)C * C * 2);                 // 0.5 MB
  bf16*  qT    = (bf16*)carve((size_t)BATCH * H * D * NTOK * 2);  // 32 MB
  bf16*  kT    = (bf16*)carve((size_t)BATCH * H * D * NTOK * 2);  // 32 MB
  bf16*  vmat  = (bf16*)carve((size_t)MTOK * C * 2);              // 32 MB
  bf16*  attn  = (bf16*)carve((size_t)BATCH * H * D * D * 2);     // 0.5 MB
  float* part  = (float*)carve((size_t)KCH * BATCH * H * D * D * 4); // 8 MB
  bf16*  y     = xb;   // x dead after GEMM1 -> reuse its slot

  cvt_f32_bf16_x4<<<(MTOK * C / 4) / 256, 256, 0, stream>>>(x, xb, MTOK * C);
  transpose_w_bf16<<<(C * C3) / 256, 256, 0, stream>>>(w_qkv, wqkvt, C, C3);
  transpose_w_bf16<<<(C * C) / 256, 256, 0, stream>>>(w_p, wpt, C, C);

  gemm_qkv<<<3072, 256, 0, stream>>>(xb, wqkvt, b_qkv, qT, kT, vmat);
  scores_partial<<<128, 256, 0, stream>>>(qT, kT, part);
  softmax_attn<<<BATCH * H, 64, 0, stream>>>(part, temp, attn);
  attn_apply<<<1024, 256, 0, stream>>>(vmat, attn, y);
  gemm_out<<<1024, 256, 0, stream>>>(y, wpt, b_p, (float*)d_out);
}
